// SwinTransformer_59407987638612
// MI455X (gfx1250) — matrix operand order repaired and checked by reading, not compiled, not run on hardware
//
#include <hip/hip_runtime.h>
#include <hip/hip_bf16.h>

typedef __bf16 bf16_t;
typedef __attribute__((ext_vector_type(16))) __bf16 v16bf;
typedef __attribute__((ext_vector_type(8)))  __bf16 v8bf;
typedef __attribute__((ext_vector_type(8)))  float  v8f;

// Build a 16-elem fragment from two contiguous 8-elem (16B) loads.
static __device__ __forceinline__ v16bf frag_cat(v8bf lo, v8bf hi) {
    return __builtin_shufflevector(lo, hi, 0,1,2,3,4,5,6,7,8,9,10,11,12,13,14,15);
}
// A-fragment: af[0..15] = A[base..base+15], base = k0 + half*16
static __device__ __forceinline__ v16bf load_afrag(const bf16_t* __restrict__ row, int k0, int half) {
    const bf16_t* p = row + k0 + half * 16;
    return frag_cat(*(const v8bf*)p, *(const v8bf*)(p + 8));
}
// B-fragment: bf[0..15] = W[kb0..kb0+15], kb0 = k0 + half*16
static __device__ __forceinline__ v16bf load_bfrag(const bf16_t* __restrict__ row, int k0, int half) {
    const bf16_t* p = row + k0 + half * 16;
    return frag_cat(*(const v8bf*)p, *(const v8bf*)(p + 8));
}

// ---------------------------------------------------------------------------
// fp32 -> bf16 convert
// ---------------------------------------------------------------------------
__global__ void k_f32_to_bf16(const float* __restrict__ in, bf16_t* __restrict__ out, int n) {
    int i = blockIdx.x * blockDim.x + threadIdx.x;
    if (i < n) out[i] = (bf16_t)in[i];
}

// ---------------------------------------------------------------------------
// Patch embed: 4x4 stride-4 conv -> [B, 3136, 96]
// ---------------------------------------------------------------------------
__global__ void k_patch_embed(const float* __restrict__ x, const float* __restrict__ w,
                              const float* __restrict__ b, float* __restrict__ out, int B) {
    long idx = (long)blockIdx.x * blockDim.x + threadIdx.x;
    long total = (long)B * 56 * 56 * 96;
    if (idx >= total) return;
    int o = idx % 96; long t = idx / 96;
    int ww = t % 56; t /= 56;
    int hh = t % 56; int bb = (int)(t / 56);
    float acc = b[o];
    for (int c = 0; c < 3; ++c)
        for (int p = 0; p < 4; ++p)
            for (int q = 0; q < 4; ++q)
                acc += x[(((long)bb*3 + c)*224 + hh*4 + p)*224 + ww*4 + q]
                     * w[((o*3 + c)*4 + p)*4 + q];
    out[((long)bb*3136 + hh*56 + ww)*96 + o] = acc;
}

// ---------------------------------------------------------------------------
// Window partition (shift roll folded in), fp32 -> bf16 windows
// ---------------------------------------------------------------------------
__global__ void k_window_partition(const float* __restrict__ x, bf16_t* __restrict__ xw,
                                   int B, int H, int W, int C, int ss) {
    long idx = (long)blockIdx.x * blockDim.x + threadIdx.x;
    long total = (long)B * H * W * C;
    if (idx >= total) return;
    int ch = idx % C; long t = idx / C;
    int n = t % 49;  long b_ = t / 49;
    int nWw = W / 7, nWin = (H / 7) * nWw;
    int b = (int)(b_ / nWin), wIdx = (int)(b_ % nWin);
    int wy = wIdx / nWw, wx = wIdx % nWw;
    int r = n / 7, c = n % 7;
    int y  = (wy*7 + r + ss) % H;
    int xc = (wx*7 + c + ss) % W;
    xw[idx] = (bf16_t)x[((long)b*H*W + (long)y*W + xc)*C + ch];
}

// ---------------------------------------------------------------------------
// Window reverse + roll back + residual add into x
// ---------------------------------------------------------------------------
__global__ void k_window_reverse_add(const float* __restrict__ p, float* __restrict__ x,
                                     int B, int H, int W, int C, int ss) {
    long idx = (long)blockIdx.x * blockDim.x + threadIdx.x;
    long total = (long)B * H * W * C;
    if (idx >= total) return;
    int ch = idx % C; long t = idx / C;
    int n = t % 49;  long b_ = t / 49;
    int nWw = W / 7, nWin = (H / 7) * nWw;
    int b = (int)(b_ / nWin), wIdx = (int)(b_ % nWin);
    int wy = wIdx / nWw, wx = wIdx % nWw;
    int r = n / 7, c = n % 7;
    int y  = (wy*7 + r + ss) % H;
    int xc = (wx*7 + c + ss) % W;
    x[((long)b*H*W + (long)y*W + xc)*C + ch] += p[idx];
}

// ---------------------------------------------------------------------------
// WMMA bf16 GEMM:  C[M,N] = A[M,K] @ W[N,K]^T (+bias)(+GELU)(+resid)
// One wave = 16x32 output (one A-frag reused over two 16x16 N tiles).
// K % 32 == 0.  flags: 1=bias, 2=gelu, 4=residual add (fp32), 8=store bf16
// ---------------------------------------------------------------------------
__global__ void __launch_bounds__(256)
k_gemm_bf16(const bf16_t* __restrict__ A, const bf16_t* __restrict__ W,
            const float* __restrict__ bias, float* __restrict__ Cf,
            bf16_t* __restrict__ Cb, int M, int K, int N, int flags) {
    int wave = (int)((blockIdx.x * blockDim.x + threadIdx.x) >> 5);
    int lane = threadIdx.x & 31;
    int tilesN  = (N + 15) >> 4;
    int tilesN2 = (tilesN + 1) >> 1;
    int tilesM  = (M + 15) >> 4;
    if (wave >= tilesM * tilesN2) return;
    int tm = wave / tilesN2, tp = wave % tilesN2;
    int tn0 = tp * 2, tn1 = tn0 + 1;
    int half = lane >> 4, mr = lane & 15;
    int am = tm*16 + mr; if (am >= M) am = M - 1;
    int bn0 = tn0*16 + mr; if (bn0 >= N) bn0 = N - 1;
    int bn1 = tn1*16 + mr; if (bn1 >= N) bn1 = N - 1;
    const bf16_t* Arow  = A + (long)am  * K;
    const bf16_t* Wrow0 = W + (long)bn0 * K;
    const bf16_t* Wrow1 = W + (long)bn1 * K;
    v8f acc0 = {0.f,0.f,0.f,0.f,0.f,0.f,0.f,0.f};
    v8f acc1 = {0.f,0.f,0.f,0.f,0.f,0.f,0.f,0.f};
    for (int k0 = 0; k0 < K; k0 += 32) {
        v16bf af  = load_afrag(Arow,  k0, half);
        v16bf bf0 = load_bfrag(Wrow0, k0, half);
        v16bf bf1 = load_bfrag(Wrow1, k0, half);
        acc0 = __builtin_amdgcn_wmma_f32_16x16x32_bf16(false, af, false, bf0,
                                                       (short)0, acc0, false, false);
        acc1 = __builtin_amdgcn_wmma_f32_16x16x32_bf16(false, af, false, bf1,
                                                       (short)0, acc1, false, false);
    }
#pragma unroll
    for (int s = 0; s < 2; ++s) {
        int cn = (tn0 + s)*16 + mr;
        if (cn >= N) continue;
        v8f acc = s ? acc1 : acc0;
        float bv = (flags & 1) ? bias[cn] : 0.f;
#pragma unroll
        for (int r = 0; r < 8; ++r) {
            int cm = tm*16 + (half ? 8 : 0) + r;
            if (cm >= M) continue;
            float v = acc[r] + bv;
            if (flags & 2) v = 0.5f * v * (1.0f + erff(v * 0.70710678118654752f));
            long off = (long)cm * N + cn;
            if (flags & 8)      Cb[off] = (bf16_t)v;
            else if (flags & 4) Cf[off] += v;
            else                Cf[off] = v;
        }
    }
}

// ---------------------------------------------------------------------------
// Fused window attention: one workgroup per (window, head).  N=49 pad 64, d=32.
// ---------------------------------------------------------------------------
__device__ __forceinline__ int region_of(int y, int D, int ss) {
    return (y < D - 7) ? 0 : ((y < D - ss) ? 1 : 2);
}

__global__ void __launch_bounds__(256)
k_window_attention(const bf16_t* __restrict__ qkv, bf16_t* __restrict__ obf,
                   const float* __restrict__ bias_table,
                   int h, int H, int W, int ss) {
    const int C = h * 32;
    int bh = blockIdx.x;
    int head = bh % h; long b_ = bh / h;
    int nWw = W / 7, nWin = (H / 7) * nWw;
    int wIdx = (int)(b_ % nWin);
    int wy = wIdx / nWw, wx = wIdx % nWw;

    __shared__ __align__(16) bf16_t qs[64*32];
    __shared__ __align__(16) bf16_t ks[64*32];
    __shared__ __align__(16) bf16_t vt[32*64];
    __shared__ __align__(16) float  Ssm[64*64];
    __shared__ __align__(16) bf16_t Ps[64*64];

    int tid = threadIdx.x;
    for (int i = tid; i < 64*32; i += 256) { qs[i] = (bf16_t)0.f; ks[i] = (bf16_t)0.f; }
    for (int i = tid; i < 32*64; i += 256) vt[i] = (bf16_t)0.f;
    __syncthreads();

    const float scale = 0.17677669529663688f;  // 32^-0.5
    for (int i = tid; i < 49*32; i += 256) {
        int n = i >> 5, dd = i & 31;
        long base = (b_*49 + n) * (long)(3*C) + head*32 + dd;
        qs[n*32 + dd] = (bf16_t)((float)qkv[base] * scale);
        ks[n*32 + dd] = qkv[base + C];
        vt[dd*64 + n] = qkv[base + 2*C];
    }
    __syncthreads();

    int wave = tid >> 5, lane = tid & 31, half = lane >> 4, mr = lane & 15;

    // ---- S = (q*scale) @ k^T + bias (+mask)  (16 tiles, 2 per wave) ----
    for (int t = wave; t < 16; t += 8) {
        int tm = t >> 2, tn = t & 3;
        v16bf af  = load_afrag(qs + (tm*16 + mr)*32, 0, half);
        v16bf bfv = load_bfrag(ks + (tn*16 + mr)*32, 0, half);
        v8f acc = {0.f,0.f,0.f,0.f,0.f,0.f,0.f,0.f};
        acc = __builtin_amdgcn_wmma_f32_16x16x32_bf16(false, af, false, bfv,
                                                      (short)0, acc, false, false);
        int j = tn*16 + mr;
#pragma unroll
        for (int r = 0; r < 8; ++r) {
            int i = tm*16 + (half ? 8 : 0) + r;
            float val = -1e30f;
            if (i < 49 && j < 49) {
                int ih = i / 7, iw = i % 7, jh = j / 7, jw = j % 7;
                int bidx = (ih - jh + 6)*13 + (iw - jw + 6);
                val = acc[r] + bias_table[bidx*h + head];
                if (ss > 0) {
                    int ri = region_of(wy*7 + ih, H, ss)*3 + region_of(wx*7 + iw, W, ss);
                    int rj = region_of(wy*7 + jh, H, ss)*3 + region_of(wx*7 + jw, W, ss);
                    if (ri != rj) val -= 100.0f;
                }
            }
            Ssm[i*64 + j] = val;
        }
    }
    __syncthreads();

    // ---- softmax: 4 lanes per row (64 rows x 4 = 256 threads) ----
    {
        int row = tid >> 2, sub = tid & 3;
        float m = -1e30f;
        if (row < 49)
            for (int j = sub; j < 49; j += 4) m = fmaxf(m, Ssm[row*64 + j]);
        m = fmaxf(m, __shfl_xor(m, 1, 32));
        m = fmaxf(m, __shfl_xor(m, 2, 32));
        float ssum = 0.f;
        if (row < 49)
            for (int j = sub; j < 49; j += 4) ssum += __expf(Ssm[row*64 + j] - m);
        ssum += __shfl_xor(ssum, 1, 32);
        ssum += __shfl_xor(ssum, 2, 32);
        float inv = 1.0f / ssum;
        for (int j = sub; j < 64; j += 4) {
            float p = (row < 49 && j < 49) ? __expf(Ssm[row*64 + j] - m) * inv : 0.f;
            Ps[row*64 + j] = (bf16_t)p;
        }
    }
    __syncthreads();

    // ---- O = P @ V  (4 m-tiles x 2 n-tiles = 8 tiles, one per wave) ----
    {
        int tm = wave >> 1, tn = wave & 1;
        v8f acc = {0.f,0.f,0.f,0.f,0.f,0.f,0.f,0.f};
#pragma unroll
        for (int k0 = 0; k0 < 64; k0 += 32) {
            v16bf af  = load_afrag(Ps + (tm*16 + mr)*64, k0, half);
            v16bf bfv = load_bfrag(vt + (tn*16 + mr)*64, k0, half);
            acc = __builtin_amdgcn_wmma_f32_16x16x32_bf16(false, af, false, bfv,
                                                          (short)0, acc, false, false);
        }
        int dd = tn*16 + mr;  // 0..31
#pragma unroll
        for (int r = 0; r < 8; ++r) {
            int i = tm*16 + (half ? 8 : 0) + r;
            if (i < 49)
                obf[(b_*49 + i) * (long)C + head*32 + dd] = (bf16_t)acc[r];
        }
    }
}

// ---------------------------------------------------------------------------
// LayerNorm: one wave per row.  Optional bf16 and/or fp32 outputs.
// ---------------------------------------------------------------------------
__global__ void __launch_bounds__(256)
k_layernorm(const float* __restrict__ x, const float* __restrict__ g,
            const float* __restrict__ b, bf16_t* __restrict__ ybf,
            float* __restrict__ yf, long rows, int C) {
    long row = (long)blockIdx.x * 8 + (threadIdx.x >> 5);
    int lane = threadIdx.x & 31;
    if (row >= rows) return;
    const float* xr = x + row * C;
    float s = 0.f, sq = 0.f;
    for (int c = lane; c < C; c += 32) { float v = xr[c]; s += v; sq += v*v; }
#pragma unroll
    for (int o = 16; o > 0; o >>= 1) {
        s  += __shfl_xor(s,  o, 32);
        sq += __shfl_xor(sq, o, 32);
    }
    float mu  = s / C;
    float var = sq / C - mu * mu;
    float inv = rsqrtf(var + 1e-5f);
    for (int c = lane; c < C; c += 32) {
        float v = (xr[c] - mu) * inv * g[c] + b[c];
        if (ybf) ybf[row * C + c] = (bf16_t)v;
        if (yf)  yf[row * C + c]  = v;
    }
}

// ---------------------------------------------------------------------------
// Patch merging gather + LayerNorm -> bf16 [B*(H/2)*(W/2), 4C]
// ---------------------------------------------------------------------------
__global__ void __launch_bounds__(256)
k_merge_ln(const float* __restrict__ x, const float* __restrict__ g,
           const float* __restrict__ b, bf16_t* __restrict__ out,
           int B, int H, int W, int C) {
    int H2 = H/2, W2 = W/2, C4 = 4*C;
    long row = blockIdx.x;               // over B*H2*W2
    int ww = row % W2; long t = row / W2;
    int hh = t % H2;  int bb = (int)(t / H2);
    __shared__ float buf[1536];
    __shared__ float r1[256], r2[256];
    int tid = threadIdx.x;
    for (int i = tid; i < C4; i += 256) {
        int q = i / C, c = i % C;
        int dy = q & 1, dx = q >> 1;     // q: 0=(0,0) 1=(1,0) 2=(0,1) 3=(1,1)
        buf[i] = x[(((long)bb*H + (2*hh + dy))*W + (2*ww + dx))*C + c];
    }
    __syncthreads();
    float s = 0.f, sq = 0.f;
    for (int i = tid; i < C4; i += 256) { float v = buf[i]; s += v; sq += v*v; }
    r1[tid] = s; r2[tid] = sq; __syncthreads();
    for (int o = 128; o > 0; o >>= 1) {
        if (tid < o) { r1[tid] += r1[tid + o]; r2[tid] += r2[tid + o]; }
        __syncthreads();
    }
    float mu  = r1[0] / C4;
    float var = r2[0] / C4 - mu * mu;
    float inv = rsqrtf(var + 1e-5f);
    for (int i = tid; i < C4; i += 256)
        out[row * (long)C4 + i] = (bf16_t)((buf[i] - mu) * inv * g[i] + b[i]);
}

// ---------------------------------------------------------------------------
// Mean pool over tokens -> bf16 [B, C]
// ---------------------------------------------------------------------------
__global__ void k_pool(const float* __restrict__ x, bf16_t* __restrict__ out,
                       int B, int L, int C) {
    int i = blockIdx.x * blockDim.x + threadIdx.x;
    if (i >= B * C) return;
    int c = i % C, b = i / C;
    float s = 0.f;
    for (int l = 0; l < L; ++l) s += x[((long)b*L + l)*C + c];
    out[i] = (bf16_t)(s / (float)L);
}

// ===========================================================================
// Host orchestration
// ===========================================================================
extern "C" void kernel_launch(void* const* d_in, const int* in_sizes, int n_in,
                              void* d_out, int out_size, void* d_ws, size_t ws_size,
                              hipStream_t stream) {
    (void)in_sizes; (void)n_in; (void)out_size; (void)ws_size;
    const int B = 32;
    const int dims_s[4]   = {96, 192, 384, 768};
    const int heads_s[4]  = {3, 6, 12, 24};
    const int depths_s[4] = {2, 2, 6, 2};

    // --- input pointers: JAX pytree flatten order (dict keys sorted) ---
    auto F = [&](int i) { return (const float*)d_in[i]; };
    const float* head_b   = F(0);
    const float* head_w   = F(1);
    const float* mg_nb[3] = {F(2), F(5), F(8)};
    const float* mg_ng[3] = {F(3), F(6), F(9)};
    const float* mg_rw[3] = {F(4), F(7), F(10)};
    const float* norm_b   = F(11);
    const float* norm_g   = F(12);
    const float* patch_b  = F(13);
    const float* patch_w  = F(14);
    struct Blk { const float *bias_table,*mlp_b1,*mlp_b2,*mlp_w1,*mlp_w2,
                             *norm2_b,*norm2_g,*proj_b,*proj_w,*qkv_b,*qkv_w; };
    Blk blk[12];
    for (int i = 0; i < 12; ++i) {
        int base = 15 + i * 11;
        blk[i] = { F(base+0), F(base+1), F(base+2), F(base+3), F(base+4),
                   F(base+5), F(base+6), F(base+7), F(base+8), F(base+9), F(base+10) };
    }
    const float* x_in = F(15 + 12*11);   // index 147

    // --- workspace bump allocator ---
    char* wsp = (char*)d_ws;
    size_t off = 0;
    auto alloc = [&](size_t bytes) -> void* {
        void* p = wsp + off;
        off += (bytes + 255) & ~(size_t)255;
        return p;
    };
    const long NE = (long)B * 3136 * 96;            // 9,633,792 (stage-0 act elems)
    float*  X    = (float*)  alloc(NE * 4);
    bf16_t* XW   = (bf16_t*) alloc(NE * 2);
    bf16_t* QKV  = (bf16_t*) alloc(3 * NE * 2);
    bf16_t* OBF  = (bf16_t*) alloc(NE * 2);
    float*  PROJ = (float*)  alloc(NE * 4);
    bf16_t* YBF  = (bf16_t*) alloc(NE * 2);
    bf16_t* HBF  = (bf16_t*) alloc(4 * NE * 2);
    bf16_t* POOL = (bf16_t*) alloc((size_t)B * 768 * 2);

    auto conv = [&](const float* src, long n) -> bf16_t* {
        bf16_t* dst = (bf16_t*)alloc((size_t)n * 2);
        k_f32_to_bf16<<<dim3((unsigned)((n + 255) / 256)), dim3(256), 0, stream>>>(src, dst, (int)n);
        return dst;
    };

    bf16_t *qkv_wb[12], *proj_wb[12], *mlp1_wb[12], *mlp2_wb[12], *red_wb[3];
    {
        int bi = 0;
        for (int s = 0; s < 4; ++s) {
            long C = dims_s[s];
            for (int j = 0; j < depths_s[s]; ++j, ++bi) {
                qkv_wb[bi]  = conv(blk[bi].qkv_w,  3*C*C);
                proj_wb[bi] = conv(blk[bi].proj_w, C*C);
                mlp1_wb[bi] = conv(blk[bi].mlp_w1, 4*C*C);
                mlp2_wb[bi] = conv(blk[bi].mlp_w2, 4*C*C);
            }
            if (s < 3) red_wb[s] = conv(mg_rw[s], (2*C)*(4*C));
        }
    }
    bf16_t* head_wb = conv(head_w, 1000L * 768);

    auto gemm = [&](const bf16_t* A, const bf16_t* Wt, const float* bias,
                    float* Cf, bf16_t* Cb, long M, int K, int N, int flags) {
        long tilesN2 = (((N + 15) / 16) + 1) / 2;
        long waves = ((M + 15) / 16) * tilesN2;
        long blocks = (waves + 7) / 8;
        k_gemm_bf16<<<dim3((unsigned)blocks), dim3(256), 0, stream>>>(
            A, Wt, bias, Cf, Cb, (int)M, K, N, flags);
    };
    auto g1 = [](long n) { return dim3((unsigned)((n + 255) / 256)); };

    // --- patch embed ---
    k_patch_embed<<<g1(NE), dim3(256), 0, stream>>>(x_in, patch_w, patch_b, X, B);

    int H = 56, W = 56, bi = 0;
    for (int s = 0; s < 4; ++s) {
        int C = dims_s[s], h = heads_s[s];
        long M = (long)B * H * W;           // tokens
        for (int j = 0; j < depths_s[s]; ++j, ++bi) {
            int ssv = (j % 2 == 1 && H > 7 && W > 7) ? 3 : 0;
            long total = M * C;
            // window partition (norm1 intentionally absent, faithful to source)
            k_window_partition<<<g1(total), dim3(256), 0, stream>>>(X, XW, B, H, W, C, ssv);
            // qkv
            gemm(XW, qkv_wb[bi], blk[bi].qkv_b, nullptr, QKV, M, C, 3*C, 1 | 8);
            // attention (fused S + softmax + P@V)
            long nAttn = (M / 49) * h;
            k_window_attention<<<dim3((unsigned)nAttn), dim3(256), 0, stream>>>(
                QKV, OBF, blk[bi].bias_table, h, H, W, ssv);
            // proj
            gemm(OBF, proj_wb[bi], blk[bi].proj_b, PROJ, nullptr, M, C, C, 1);
            // reverse + residual into X
            k_window_reverse_add<<<g1(total), dim3(256), 0, stream>>>(PROJ, X, B, H, W, C, ssv);
            // norm2 -> bf16
            k_layernorm<<<dim3((unsigned)((M + 7) / 8)), dim3(256), 0, stream>>>(
                X, blk[bi].norm2_g, blk[bi].norm2_b, YBF, (float*)nullptr, M, C);
            // MLP
            gemm(YBF, mlp1_wb[bi], blk[bi].mlp_b1, nullptr, HBF, M, C, 4*C, 1 | 2 | 8);
            gemm(HBF, mlp2_wb[bi], blk[bi].mlp_b2, X, nullptr, M, 4*C, C, 1 | 4);
        }
        if (s < 3) {
            long rows = (long)B * (H/2) * (W/2);
            k_merge_ln<<<dim3((unsigned)rows), dim3(256), 0, stream>>>(
                X, mg_ng[s], mg_nb[s], YBF, B, H, W, C);
            gemm(YBF, red_wb[s], nullptr, X, nullptr, rows, 4*C, 2*C, 0);
            H /= 2; W /= 2;
        }
    }

    // --- final LN -> mean pool -> head ---
    {
        long M = (long)B * 49;   // H=W=7, C=768
        k_layernorm<<<dim3((unsigned)((M + 7) / 8)), dim3(256), 0, stream>>>(
            X, norm_g, norm_b, (bf16_t*)nullptr, PROJ, M, 768);
        k_pool<<<g1((long)B * 768), dim3(256), 0, stream>>>(PROJ, POOL, B, 49, 768);
        gemm(POOL, head_wb, head_b, (float*)d_out, nullptr, B, 768, 1000, 1);
    }
}
